// RITS_71846212927904
// MI455X (gfx1250) — matrix operand order repaired and checked by reading, not compiled, not run on hardware
//
#include <hip/hip_runtime.h>
#include <hip/hip_bf16.h>
#include <cmath>

// Problem dims (fixed by reference)
#define B_DIM 4096
#define L_DIM 96
#define N_DIM 64
#define H_DIM 128
#define LN_STRIDE (L_DIM * N_DIM)   // 6144

typedef __attribute__((ext_vector_type(16))) _Float16 v16h;
typedef __attribute__((ext_vector_type(8)))  _Float16 v8h;
typedef __attribute__((ext_vector_type(8)))  float    v8f;

// f16 weight layout inside workspace / LDS (offsets in halves)
#define OFF_WGH   0        // W_gh   [128 x  64]  -> 8192
#define OFF_WHIST 8192     // W_hist [ 64 x 128]  -> 8192
#define OFF_WFEAT 16384    // W_feat [ 64 x  64]  -> 4096 (diag zeroed)
#define OFF_WCOMB 20480    // W_comb [ 64 x 128]  -> 8192
#define OFF_WIH   28672    // W_ih   [512 x 128]  -> 65536
#define OFF_WHH   94208    // W_hh   [512 x 128]  -> 65536
#define WTOTAL    159744   // halves = 319488 bytes
#define OFF_F32WS_BYTES (WTOTAL * 2)   // f32 region: invDenom[96]

// ---------------------------------------------------------------------------
// Fragment loaders (CDNA5 wave32 WMMA f16 16x16x32 layouts, CDNA5 ISA)
// A (16xK slice, row-major f16 in LDS): lane ln holds row m=ln; two contiguous
// 8-half chunks at k = kbase + hi*8 and kbase + 16 + hi*8.
// B (W row-major [out x K] f16): lane holds out-row colBase+ln, two contiguous
// 8-half chunks at k = kbase + hi*8 and kbase + 16 + hi*8 -> weights need no transpose.
// ---------------------------------------------------------------------------
__device__ __forceinline__ v16h load_a_frag(const _Float16* base, int stride,
                                            int kbase, int ln, int hi) {
  const _Float16* p = base + ln * stride + kbase + hi * 8;
  v8h a0 = *(const v8h*)p;
  v8h a1 = *(const v8h*)(p + 16);
  v16h r;
#pragma unroll
  for (int i = 0; i < 8; ++i) { r[i] = a0[i]; r[i + 8] = a1[i]; }
  return r;
}

__device__ __forceinline__ v16h load_b_frag(const _Float16* W, int stride,
                                            int colBase, int kbase, int ln, int hi) {
  const _Float16* p = W + (size_t)(colBase + ln) * stride + kbase + hi * 8;
  v8h b0 = *(const v8h*)p;
  v8h b1 = *(const v8h*)(p + 16);
  v16h r;
#pragma unroll
  for (int i = 0; i < 8; ++i) { r[i] = b0[i]; r[i + 8] = b1[i]; }
  return r;
}

#define WMMA_F16(A, B, C) \
  __builtin_amdgcn_wmma_f32_16x16x32_f16(false, (A), false, (B), (short)0, (C), false, false)

__device__ __forceinline__ float sigmoidf_(float x) {
  return 1.0f / (1.0f + __expf(-x));
}

// ---------------------------------------------------------------------------
// Pre-pass: f32 -> f16 weight conversion (optionally zeroing the diagonal,
// used by FeatureRegression's W_feat * (1 - I)).
// ---------------------------------------------------------------------------
__global__ void cvt_f16_kernel(const float* __restrict__ src, _Float16* __restrict__ dst,
                               int count, int diagN) {
  int i = blockIdx.x * 256 + threadIdx.x;
  if (i < count) {
    float v = src[i];
    if (diagN && (i / diagN) == (i % diagN)) v = 0.0f;
    dst[i] = (_Float16)v;
  }
}

// ---------------------------------------------------------------------------
// Pre-pass: per-step global mask sum -> invDenom[t] = 1/(sum+1e-5).
// Also zero-initializes the loss accumulator (deterministic per launch).
// ---------------------------------------------------------------------------
__global__ void denom_kernel(const float* __restrict__ masks,
                             float* __restrict__ invDenom,
                             float* __restrict__ loss_out) {
  int t = blockIdx.x;
  __shared__ float red[256];
  float s = 0.0f;
  for (int i = threadIdx.x; i < B_DIM * N_DIM; i += 256) {
    int b = i >> 6;
    int n = i & 63;
    s += masks[(size_t)b * LN_STRIDE + (size_t)t * N_DIM + n];
  }
  red[threadIdx.x] = s;
  __syncthreads();
  for (int o = 128; o > 0; o >>= 1) {
    if (threadIdx.x < o) red[threadIdx.x] += red[threadIdx.x + o];
    __syncthreads();
  }
  if (threadIdx.x == 0) invDenom[t] = 1.0f / (red[0] + 1e-5f);
  if (t == 0 && threadIdx.x == 0) *loss_out = 0.0f;
}

// ---------------------------------------------------------------------------
// Main persistent-state kernel: one block = 16 batch rows, 4 waves, all
// weights resident in LDS (312KB) + 8KB activation staging = 320KB.
// ---------------------------------------------------------------------------
__global__ __launch_bounds__(128, 1)
void rits_main(const float* __restrict__ xt, const float* __restrict__ masks,
               const float* __restrict__ deltas,
               const float* __restrict__ b_gh, const float* __restrict__ W_gx,
               const float* __restrict__ b_gx, const float* __restrict__ b_hist,
               const float* __restrict__ b_feat, const float* __restrict__ b_comb,
               const float* __restrict__ b_ih, const float* __restrict__ b_hh,
               const _Float16* __restrict__ wsh, const float* __restrict__ invDenom,
               float* __restrict__ out_imp, float* __restrict__ out_loss) {
  __shared__ __align__(16) _Float16 s_w[WTOTAL];   // all weights, f16
  __shared__ __align__(16) _Float16 s_h[16 * 128]; // decayed h tile (A source)
  __shared__ __align__(16) _Float16 s_act[16 * 128]; // staging for A operands

  const int tid = threadIdx.x;

  // Cooperative weight preload: global -> LDS as 16B chunks.
  {
    const uint4* src = (const uint4*)wsh;
    uint4* dst = (uint4*)s_w;
    for (int i = tid; i < WTOTAL / 8; i += 128) dst[i] = src[i];
  }
  __syncthreads();

  const int lane = tid & 31;
  const int w    = tid >> 5;        // wave id 0..3
  const int ln   = lane & 15;
  const int hi   = lane >> 4;
  const int nx   = 16 * w + ln;     // owned N-space column (0..63)
  const int n1   = nx;              // owned H-space columns
  const int n2   = 64 + nx;
  const int rbase = 8 * hi;         // owned rows rbase..rbase+7 (D-frag layout)
  const int b0   = blockIdx.x * 16; // batch tile base

  // Per-thread scalar parameters for the owned columns.
  const float gxd   = W_gx[nx * N_DIM + nx];
  const float bgx   = b_gx[nx];
  const float bhist = b_hist[nx];
  const float bfeat = b_feat[nx];
  const float bcomb = b_comb[nx];
  const float bgh[2] = { b_gh[n1], b_gh[n2] };
  float bg[4][2];
#pragma unroll
  for (int g = 0; g < 4; ++g) {
    bg[g][0] = b_ih[128 * g + n1] + b_hh[128 * g + n1];
    bg[g][1] = b_ih[128 * g + n2] + b_hh[128 * g + n2];
  }

  // Recurrent state in registers (D-frag layout: [hcol half j][row v]).
  float h[2][8], c[2][8];
#pragma unroll
  for (int j = 0; j < 2; ++j)
#pragma unroll
    for (int v = 0; v < 8; ++v) { h[j][v] = 0.0f; c[j][v] = 0.0f; }

  float lsum = 0.0f;

  for (int t = 0; t < L_DIM; ++t) {
    // ---- load x, m, d for owned (row, col) elements ----
    float xr[8], mr[8], dr[8];
#pragma unroll
    for (int v = 0; v < 8; ++v) {
      size_t idx = (size_t)(b0 + rbase + v) * LN_STRIDE + (size_t)t * N_DIM + nx;
      float x = xt[idx];
      xr[v] = (x == x) ? x : 0.0f;     // nan_to_num
      mr[v] = masks[idx];
      dr[v] = deltas[idx];
    }

    __syncthreads();                    // prior-step consumers of s_act/s_h done
    // stage d as A [16 x 64]
#pragma unroll
    for (int v = 0; v < 8; ++v) s_act[(rbase + v) * 64 + nx] = (_Float16)dr[v];
    __syncthreads();

    // ---- gamma_h = exp(-relu(d @ W_gh.T + b_gh)); h *= gamma_h ----
#pragma unroll
    for (int j = 0; j < 2; ++j) {
      const int colBase = 16 * (w + 4 * j);
      v8f acc = {};
#pragma unroll
      for (int k = 0; k < 64; k += 32)
        acc = WMMA_F16(load_a_frag(s_act, 64, k, ln, hi),
                       load_b_frag(s_w + OFF_WGH, 64, colBase, k, ln, hi), acc);
#pragma unroll
      for (int v = 0; v < 8; ++v) {
        float g = acc[v] + bgh[j];
        g = g > 0.0f ? g : 0.0f;
        h[j][v] *= __expf(-g);
      }
    }

    // stage decayed h as A [16 x 128] (stays valid through the gate matmuls)
#pragma unroll
    for (int j = 0; j < 2; ++j)
#pragma unroll
      for (int v = 0; v < 8; ++v)
        s_h[(rbase + v) * 128 + nx + 64 * j] = (_Float16)h[j][v];
    __syncthreads();

    // ---- x_h = h @ W_hist.T + b_hist ----
    float xh[8], xc[8];
    float l1 = 0.0f, l2 = 0.0f, l3 = 0.0f;
    {
      v8f acc = {};
#pragma unroll
      for (int k = 0; k < 128; k += 32)
        acc = WMMA_F16(load_a_frag(s_h, 128, k, ln, hi),
                       load_b_frag(s_w + OFF_WHIST, 128, 16 * w, k, ln, hi), acc);
#pragma unroll
      for (int v = 0; v < 8; ++v) {
        xh[v] = acc[v] + bhist;
        l1 += fabsf(xr[v] - xh[v]) * mr[v];
        xc[v] = mr[v] * xr[v] + (1.0f - mr[v]) * xh[v];
      }
    }

    // stage x_c as A [16 x 64] (gamma_h reads of s_act retired at last barrier)
#pragma unroll
    for (int v = 0; v < 8; ++v) s_act[(rbase + v) * 64 + nx] = (_Float16)xc[v];
    __syncthreads();

    // ---- z_h = x_c @ (W_feat*(1-I)).T + b_feat ----
    float zh[8];
    {
      v8f acc = {};
#pragma unroll
      for (int k = 0; k < 64; k += 32)
        acc = WMMA_F16(load_a_frag(s_act, 64, k, ln, hi),
                       load_b_frag(s_w + OFF_WFEAT, 64, 16 * w, k, ln, hi), acc);
#pragma unroll
      for (int v = 0; v < 8; ++v) {
        zh[v] = acc[v] + bfeat;
        l2 += fabsf(xr[v] - zh[v]) * mr[v];
      }
    }

    // ---- gamma_x (elementwise diag decay) ----
    float gx[8];
#pragma unroll
    for (int v = 0; v < 8; ++v) {
      float g = dr[v] * gxd + bgx;
      g = g > 0.0f ? g : 0.0f;
      gx[v] = __expf(-g);
    }

    __syncthreads();                    // z_h A-reads done
    // stage concat(gamma_x, m) as A [16 x 128]
#pragma unroll
    for (int v = 0; v < 8; ++v) {
      s_act[(rbase + v) * 128 + nx]      = (_Float16)gx[v];
      s_act[(rbase + v) * 128 + 64 + nx] = (_Float16)mr[v];
    }
    __syncthreads();

    // ---- alpha = concat(gamma_x, m) @ W_comb.T + b_comb; combine; imputation ----
    float cc[8];
    {
      v8f acc = {};
#pragma unroll
      for (int k = 0; k < 128; k += 32)
        acc = WMMA_F16(load_a_frag(s_act, 128, k, ln, hi),
                       load_b_frag(s_w + OFF_WCOMB, 128, 16 * w, k, ln, hi), acc);
#pragma unroll
      for (int v = 0; v < 8; ++v) {
        float al = acc[v] + bcomb;
        float ch = al * zh[v] + (1.0f - al) * xh[v];
        l3 += fabsf(xr[v] - ch) * mr[v];
        cc[v] = mr[v] * xr[v] + (1.0f - mr[v]) * ch;
        size_t idx = (size_t)(b0 + rbase + v) * LN_STRIDE + (size_t)t * N_DIM + nx;
        out_imp[idx] = cc[v];
      }
    }

    __syncthreads();                    // alpha A-reads done
    // stage concat(c_c, m) as A [16 x 128]
#pragma unroll
    for (int v = 0; v < 8; ++v) {
      s_act[(rbase + v) * 128 + nx]      = (_Float16)cc[v];
      s_act[(rbase + v) * 128 + 64 + nx] = (_Float16)mr[v];
    }
    __syncthreads();

    // ---- gates = concat(c_c,m) @ W_ih.T + h @ W_hh.T + (b_ih+b_hh) ----
    // wave w owns gate tiles 8g+4j+w -> exactly the h/c columns it holds.
    float gr[4][2][8];
#pragma unroll
    for (int g = 0; g < 4; ++g) {
#pragma unroll
      for (int j = 0; j < 2; ++j) {
        const int colBase = 16 * (8 * g + 4 * j + w);
        v8f acc = {};
#pragma unroll
        for (int k = 0; k < 128; k += 32)
          acc = WMMA_F16(load_a_frag(s_act, 128, k, ln, hi),
                         load_b_frag(s_w + OFF_WIH, 128, colBase, k, ln, hi), acc);
#pragma unroll
        for (int k = 0; k < 128; k += 32)
          acc = WMMA_F16(load_a_frag(s_h, 128, k, ln, hi),
                         load_b_frag(s_w + OFF_WHH, 128, colBase, k, ln, hi), acc);
#pragma unroll
        for (int v = 0; v < 8; ++v) gr[g][j][v] = acc[v] + bg[g][j];
      }
    }

    // ---- LSTM cell update (pure register math, layouts aligned) ----
#pragma unroll
    for (int j = 0; j < 2; ++j)
#pragma unroll
      for (int v = 0; v < 8; ++v) {
        float ig = sigmoidf_(gr[0][j][v]);
        float fg = sigmoidf_(gr[1][j][v]);
        float gg = tanhf(gr[2][j][v]);
        float og = sigmoidf_(gr[3][j][v]);
        c[j][v] = fg * c[j][v] + ig * gg;
        h[j][v] = og * tanhf(c[j][v]);
      }

    lsum += (l1 + l2 + l3) * invDenom[t];
  }

  // ---- loss reduction: wave shfl-reduce then one atomic per wave ----
  float v = lsum;
#pragma unroll
  for (int o = 16; o > 0; o >>= 1) v += __shfl_down(v, o, 32);
  if (lane == 0) atomicAdd(out_loss, v);
}

// ---------------------------------------------------------------------------
extern "C" void kernel_launch(void* const* d_in, const int* in_sizes, int n_in,
                              void* d_out, int out_size, void* d_ws, size_t ws_size,
                              hipStream_t stream) {
  const float* xt     = (const float*)d_in[0];
  const float* masks  = (const float*)d_in[1];
  const float* deltas = (const float*)d_in[2];
  const float* W_gh   = (const float*)d_in[3];
  const float* b_gh   = (const float*)d_in[4];
  const float* W_gx   = (const float*)d_in[5];
  const float* b_gx   = (const float*)d_in[6];
  const float* W_hist = (const float*)d_in[7];
  const float* b_hist = (const float*)d_in[8];
  const float* W_feat = (const float*)d_in[9];
  const float* b_feat = (const float*)d_in[10];
  const float* W_comb = (const float*)d_in[11];
  const float* b_comb = (const float*)d_in[12];
  const float* W_ih   = (const float*)d_in[13];
  const float* W_hh   = (const float*)d_in[14];
  const float* b_ih   = (const float*)d_in[15];
  const float* b_hh   = (const float*)d_in[16];

  _Float16* wsh = (_Float16*)d_ws;
  float* invDenom = (float*)((char*)d_ws + OFF_F32WS_BYTES);

  float* out_imp  = (float*)d_out;
  float* out_loss = out_imp + (size_t)B_DIM * L_DIM * N_DIM;

  // f32 -> f16 weight conversion (W_feat diagonal zeroed per reference).
  cvt_f16_kernel<<<(H_DIM * N_DIM + 255) / 256, 256, 0, stream>>>(W_gh,   wsh + OFF_WGH,   H_DIM * N_DIM, 0);
  cvt_f16_kernel<<<(N_DIM * H_DIM + 255) / 256, 256, 0, stream>>>(W_hist, wsh + OFF_WHIST, N_DIM * H_DIM, 0);
  cvt_f16_kernel<<<(N_DIM * N_DIM + 255) / 256, 256, 0, stream>>>(W_feat, wsh + OFF_WFEAT, N_DIM * N_DIM, N_DIM);
  cvt_f16_kernel<<<(N_DIM * 2 * N_DIM + 255) / 256, 256, 0, stream>>>(W_comb, wsh + OFF_WCOMB, N_DIM * 2 * N_DIM, 0);
  cvt_f16_kernel<<<(4 * H_DIM * 2 * N_DIM + 255) / 256, 256, 0, stream>>>(W_ih, wsh + OFF_WIH, 4 * H_DIM * 2 * N_DIM, 0);
  cvt_f16_kernel<<<(4 * H_DIM * H_DIM + 255) / 256, 256, 0, stream>>>(W_hh, wsh + OFF_WHH, 4 * H_DIM * H_DIM, 0);

  // Per-step global mask denominators + loss zero-init.
  denom_kernel<<<96, 256, 0, stream>>>(masks, invDenom, out_loss);

  // Main recurrent kernel: 256 blocks x 128 threads (16 batch rows / block).
  rits_main<<<B_DIM / 16, 128, 0, stream>>>(
      xt, masks, deltas, b_gh, W_gx, b_gx, b_hist, b_feat, b_comb,
      b_ih, b_hh, wsh, invDenom, out_imp, out_loss);
}
